// GATSBYGene_80908593922532
// MI455X (gfx1250) — matrix operand order repaired and checked by reading, not compiled, not run on hardware
//
#include <hip/hip_runtime.h>
#include <cstdint>
#include <cstddef>

// Problem constants (match reference)
#define NN  50000
#define DD  256
#define HH  8
#define DHH 32
#define EE  800000

typedef __attribute__((ext_vector_type(16))) _Float16     v16h;
typedef __attribute__((ext_vector_type(8)))  float        v8f;
typedef __attribute__((ext_vector_type(4)))  float        f4;
typedef __attribute__((ext_vector_type(4)))  unsigned int u4;

union FragH { u4 q[2]; v16h v; };

__device__ __forceinline__ unsigned short h_bits(_Float16 h) {
    return __builtin_bit_cast(unsigned short, h);
}

// ---------------------------------------------------------------------------
// One-shot weight conversion: W[k][n] fp32 -> transposed packed f16 hi/lo
// arrays whi/wlo[n][k] (ushort). 128 KB each -> stays L2/WGP$-resident and
// is loaded directly as WMMA B-fragments (no per-block conversion).
// ---------------------------------------------------------------------------
__global__ void k_convw(const float* __restrict__ W,
                        unsigned short* __restrict__ whi,
                        unsigned short* __restrict__ wlo) {
    int idx = blockIdx.x * blockDim.x + threadIdx.x;
    if (idx >= DD * DD) return;
    int k = idx >> 8, n = idx & 255;          // coalesced read of W[k][*]
    float v = W[(size_t)k * DD + n];
    _Float16 hh = (_Float16)v; float rr = v - (float)hh; _Float16 ll = (_Float16)rr;
    whi[(size_t)n * DD + k] = h_bits(hh);
    wlo[(size_t)n * DD + k] = h_bits(ll);
}

// ---------------------------------------------------------------------------
// GEMM: C[M x 256] = A[M x 256] @ W[256 x 256] (+ bias), fp32 in/out.
// A is split on the fly into f16 hi/lo via LDS (shared by the 8 waves);
// B comes pre-split/pre-transposed from global. 3x wmma_f32_16x16x32_f16
// per tile per chunk gives fp32-class accuracy at f16 matrix-core rate.
// Block: 256 threads (8 waves). Tile: 32 rows x 256 cols. K chunk = 32.
// ---------------------------------------------------------------------------
__global__ __launch_bounds__(256) void k_gemm_bias(
    const float* __restrict__ A,
    const unsigned short* __restrict__ Bhi, const unsigned short* __restrict__ Blo,
    const float* __restrict__ bias, float* __restrict__ C, int M)
{
    constexpr int SA = 40;  // padded LDS row stride (ushorts)
    __shared__ __align__(16) unsigned short sAhi[32 * SA];
    __shared__ __align__(16) unsigned short sAlo[32 * SA];

    const int tid  = threadIdx.x;
    const int lane = tid & 31;
    const int wv   = tid >> 5;           // wave id 0..7 -> col-tile pair
    const int row_base = blockIdx.x * 32;

    const v8f vzero = {0.f,0.f,0.f,0.f,0.f,0.f,0.f,0.f};
    v8f acc[2][2];
    acc[0][0] = vzero; acc[0][1] = vzero; acc[1][0] = vzero; acc[1][1] = vzero;

    const int mrow = lane & 15;
    const int ka   = (lane >> 4) * 16;   // A: lanes<16 -> K 0..15; lanes>=16 -> K 16..31
    const int kb   = (lane >> 4) * 16;   // B: lanes<16 -> K 0..15; lanes>=16 -> K 16..31
    const int col0 = (wv * 2 + 0) * 16 + mrow;
    const int col1 = (wv * 2 + 1) * 16 + mrow;

    for (int k0 = 0; k0 < DD; k0 += 32) {
        // ---- stage A chunk (32 rows x 32 k) as hi/lo f16 in LDS ----
        {
            const int r  = tid >> 3;
            const int c4 = (tid & 7) * 4;
            f4 v = {0.f, 0.f, 0.f, 0.f};
            const int gr = row_base + r;
            if (gr < M) v = *(const f4*)(A + (size_t)gr * DD + k0 + c4);
            _Float16 h0 = (_Float16)v.x; float r0 = v.x - (float)h0; _Float16 l0 = (_Float16)r0;
            _Float16 h1 = (_Float16)v.y; float r1 = v.y - (float)h1; _Float16 l1 = (_Float16)r1;
            _Float16 h2 = (_Float16)v.z; float r2 = v.z - (float)h2; _Float16 l2 = (_Float16)r2;
            _Float16 h3 = (_Float16)v.w; float r3 = v.w - (float)h3; _Float16 l3 = (_Float16)r3;
            *(unsigned int*)&sAhi[r*SA + c4    ] = (unsigned)h_bits(h0) | ((unsigned)h_bits(h1) << 16);
            *(unsigned int*)&sAhi[r*SA + c4 + 2] = (unsigned)h_bits(h2) | ((unsigned)h_bits(h3) << 16);
            *(unsigned int*)&sAlo[r*SA + c4    ] = (unsigned)h_bits(l0) | ((unsigned)h_bits(l1) << 16);
            *(unsigned int*)&sAlo[r*SA + c4 + 2] = (unsigned)h_bits(l2) | ((unsigned)h_bits(l3) << 16);
        }
        __syncthreads();

        // ---- A fragments from LDS (contiguous k = 16h + s order, same as B) ----
        FragH fah[2], fal[2], fbh[2], fbl[2];
        #pragma unroll
        for (int rt = 0; rt < 2; ++rt) {
            const unsigned short* p  = &sAhi[(rt*16 + mrow) * SA];
            fah[rt].q[0] = *(const u4*)(p + ka);
            fah[rt].q[1] = *(const u4*)(p + ka + 8);
            const unsigned short* p2 = &sAlo[(rt*16 + mrow) * SA];
            fal[rt].q[0] = *(const u4*)(p2 + ka);
            fal[rt].q[1] = *(const u4*)(p2 + ka + 8);
        }
        // ---- B fragments straight from global packed arrays (L2-resident) ----
        {
            const unsigned short* p0h = Bhi + (size_t)col0 * DD + k0 + kb;
            const unsigned short* p0l = Blo + (size_t)col0 * DD + k0 + kb;
            const unsigned short* p1h = Bhi + (size_t)col1 * DD + k0 + kb;
            const unsigned short* p1l = Blo + (size_t)col1 * DD + k0 + kb;
            fbh[0].q[0] = *(const u4*)(p0h);  fbh[0].q[1] = *(const u4*)(p0h + 8);
            fbl[0].q[0] = *(const u4*)(p0l);  fbl[0].q[1] = *(const u4*)(p0l + 8);
            fbh[1].q[0] = *(const u4*)(p1h);  fbh[1].q[1] = *(const u4*)(p1h + 8);
            fbl[1].q[0] = *(const u4*)(p1l);  fbl[1].q[1] = *(const u4*)(p1l + 8);
        }
        // ---- hi/lo compensated WMMA ----
        #pragma unroll
        for (int rt = 0; rt < 2; ++rt) {
            #pragma unroll
            for (int ci = 0; ci < 2; ++ci) {
                v8f c = acc[rt][ci];
                c = __builtin_amdgcn_wmma_f32_16x16x32_f16(false, fah[rt].v, false, fbh[ci].v, (short)0, c, false, false);
                c = __builtin_amdgcn_wmma_f32_16x16x32_f16(false, fal[rt].v, false, fbh[ci].v, (short)0, c, false, false);
                c = __builtin_amdgcn_wmma_f32_16x16x32_f16(false, fah[rt].v, false, fbl[ci].v, (short)0, c, false, false);
                acc[rt][ci] = c;
            }
        }
        __syncthreads();
    }

    // ---- store (C/D layout: VGPR i -> M = i (lanes<16) / 8+i (lanes>=16)) ----
    #pragma unroll
    for (int rt = 0; rt < 2; ++rt) {
        #pragma unroll
        for (int ci = 0; ci < 2; ++ci) {
            const int col = (wv*2 + ci) * 16 + (lane & 15);
            const float bv = bias ? bias[col] : 0.0f;
            const int rb = row_base + rt*16 + ((lane >> 4) ? 8 : 0);
            #pragma unroll
            for (int i = 0; i < 8; ++i) {
                const int r = rb + i;
                if (r < M) C[(size_t)r * DD + col] = acc[rt][ci][i] + bv;
            }
        }
    }
}

// ---------------------------------------------------------------------------
// CSR build (deterministic): histogram -> scan -> scatter -> per-segment sort
// ---------------------------------------------------------------------------
__global__ void k_zero_i32(int* p, int n) {
    int i = blockIdx.x * blockDim.x + threadIdx.x;
    if (i < n) p[i] = 0;
}

__global__ void k_hist(const int* __restrict__ dstv, int* __restrict__ cnt) {
    int e = blockIdx.x * blockDim.x + threadIdx.x;
    if (e < EE) atomicAdd(&cnt[dstv[e]], 1);
}

__global__ __launch_bounds__(1024) void k_scan(const int* __restrict__ cnt,
                                               int* __restrict__ offs,
                                               int* __restrict__ cur) {
    __shared__ int sd[1024];
    const int t = threadIdx.x;
    int run = 0;
    for (int base = 0; base < NN; base += 1024) {
        int v = (base + t < NN) ? cnt[base + t] : 0;
        sd[t] = v;
        __syncthreads();
        for (int off = 1; off < 1024; off <<= 1) {
            int x = (t >= off) ? sd[t - off] : 0;
            __syncthreads();
            sd[t] += x;
            __syncthreads();
        }
        int excl  = run + sd[t] - v;
        int total = sd[1023];
        if (base + t < NN) { offs[base + t] = excl; cur[base + t] = excl; }
        run += total;
        __syncthreads();
    }
    if (t == 0) offs[NN] = run;
}

__global__ void k_scatter(const int* __restrict__ srcv, const int* __restrict__ dstv,
                          int* __restrict__ cur, int* __restrict__ csr) {
    int e = blockIdx.x * blockDim.x + threadIdx.x;
    if (e < EE) {
        int p = atomicAdd(&cur[dstv[e]], 1);
        csr[p] = srcv[e];
    }
}

// Sort each node's src list so FP accumulation order is deterministic.
__global__ void k_sort(const int* __restrict__ offs, int* __restrict__ csr) {
    int n = blockIdx.x * blockDim.x + threadIdx.x;
    if (n >= NN) return;
    int lo = offs[n], hi = offs[n + 1];
    for (int i = lo + 1; i < hi; ++i) {
        int key = csr[i];
        int j = i - 1;
        while (j >= lo && csr[j] > key) { csr[j + 1] = csr[j]; --j; }
        csr[j + 1] = key;
    }
}

// ---------------------------------------------------------------------------
// al_src[n,h] = <h[n,h,:], a_src[h,:]>, al_dst likewise. Thread per (n,h).
// ---------------------------------------------------------------------------
__global__ void k_alproj(const float* __restrict__ hmat, const float* __restrict__ asrc,
                         const float* __restrict__ adst, float* __restrict__ als,
                         float* __restrict__ ald) {
    int idx = blockIdx.x * blockDim.x + threadIdx.x;
    if (idx >= NN * HH) return;
    int n = idx >> 3, hh = idx & 7;
    const float* hr = hmat + (size_t)n * DD + hh * DHH;
    const float* as = asrc + hh * DHH;
    const float* ad = adst + hh * DHH;
    float s1 = 0.f, s2 = 0.f;
    #pragma unroll 8
    for (int d = 0; d < DHH; ++d) { float v = hr[d]; s1 += v * as[d]; s2 += v * ad[d]; }
    als[idx] = s1;
    ald[idx] = s2;
}

// ---------------------------------------------------------------------------
// Segment softmax weights per (dst node, head): max, then w=exp(e-max), denom.
// Thread per (n,h); no atomics -> deterministic.
// ---------------------------------------------------------------------------
__global__ void k_attn(const int* __restrict__ offs, const int* __restrict__ csr,
                       const float* __restrict__ als, const float* __restrict__ ald,
                       float* __restrict__ wbuf, float* __restrict__ denom) {
    int idx = blockIdx.x * blockDim.x + threadIdx.x;
    if (idx >= NN * HH) return;
    int n = idx >> 3, hh = idx & 7;
    int lo = offs[n], hi = offs[n + 1];
    float ad = ald[idx];
    float m = -1e30f;
    for (int i = lo; i < hi; ++i) {
        float v = als[csr[i] * HH + hh] + ad;
        v = v > 0.f ? v : 0.2f * v;              // leaky_relu(0.2)
        m = fmaxf(m, v);
    }
    float sum = 0.f;
    for (int i = lo; i < hi; ++i) {
        float v = als[csr[i] * HH + hh] + ad;
        v = v > 0.f ? v : 0.2f * v;
        float w = __expf(v - m);
        wbuf[(size_t)i * HH + hh] = w;
        sum += w;
    }
    denom[idx] = sum;
}

// ---------------------------------------------------------------------------
// Aggregation: wave per dst node. acc[j] (j=head) holds channel 32*j+lane.
// out = ELU(sum_e alpha_e * h[src_e]). All gathers hit L2 (h is 51 MB).
// ---------------------------------------------------------------------------
__global__ __launch_bounds__(256) void k_agg(const int* __restrict__ offs,
                                             const int* __restrict__ csr,
                                             const float* __restrict__ wbuf,
                                             const float* __restrict__ denom,
                                             const float* __restrict__ hmat,
                                             float* __restrict__ gout) {
    int gw   = (blockIdx.x * blockDim.x + threadIdx.x) >> 5;  // node
    int lane = threadIdx.x & 31;
    if (gw >= NN) return;
    float rd[8];
    #pragma unroll
    for (int j = 0; j < 8; ++j) rd[j] = 1.0f / (denom[gw * HH + j] + 1e-16f);
    float acc[8] = {0.f,0.f,0.f,0.f,0.f,0.f,0.f,0.f};
    int lo = offs[gw], hi = offs[gw + 1];
    for (int i = lo; i < hi; ++i) {
        int s = csr[i];
        const float* hr = hmat + (size_t)s * DD;
        const float* wr = wbuf + (size_t)i * HH;
        if (i + 1 < hi) __builtin_prefetch(hmat + (size_t)csr[i + 1] * DD, 0, 1);
        #pragma unroll
        for (int j = 0; j < 8; ++j) acc[j] += (wr[j] * rd[j]) * hr[j * 32 + lane];
    }
    float* go = gout + (size_t)gw * DD;
    #pragma unroll
    for (int j = 0; j < 8; ++j) {
        float a = acc[j];
        go[j * 32 + lane] = a > 0.f ? a : (__expf(a) - 1.0f);  // ELU
    }
}

// ---------------------------------------------------------------------------
extern "C" void kernel_launch(void* const* d_in, const int* in_sizes, int n_in,
                              void* d_out, int out_size, void* d_ws, size_t ws_size,
                              hipStream_t stream) {
    (void)in_sizes; (void)n_in; (void)out_size; (void)ws_size;
    const float* x   = (const float*)d_in[0];
    const int*   ei  = (const int*)d_in[1];
    const float* W1  = (const float*)d_in[2];
    const float* as1 = (const float*)d_in[3];
    const float* ad1 = (const float*)d_in[4];
    const float* W2  = (const float*)d_in[5];
    const float* as2 = (const float*)d_in[6];
    const float* ad2 = (const float*)d_in[7];
    const float* Wl  = (const float*)d_in[8];
    const float* bl  = (const float*)d_in[9];
    float*       out = (float*)d_out;

    const int* srcv = ei;
    const int* dstv = ei + EE;

    // Workspace layout (~135 MB)
    char* p = (char*)d_ws;
    auto take = [&](size_t bytes) -> char* {
        char* r = p; p += (bytes + 255) & ~(size_t)255; return r;
    };
    int*   cnt   = (int*)take(sizeof(int) * NN);
    int*   offs  = (int*)take(sizeof(int) * (NN + 1));
    int*   cur   = (int*)take(sizeof(int) * NN);
    int*   csr   = (int*)take(sizeof(int) * EE);
    float* als   = (float*)take(sizeof(float) * NN * HH);
    float* ald   = (float*)take(sizeof(float) * NN * HH);
    float* denom = (float*)take(sizeof(float) * NN * HH);
    float* wbuf  = (float*)take(sizeof(float) * (size_t)EE * HH);
    float* hbuf  = (float*)take(sizeof(float) * (size_t)NN * DD);
    float* gbuf  = (float*)take(sizeof(float) * (size_t)NN * DD);
    unsigned short* w1hi = (unsigned short*)take(sizeof(unsigned short) * DD * DD);
    unsigned short* w1lo = (unsigned short*)take(sizeof(unsigned short) * DD * DD);
    unsigned short* w2hi = (unsigned short*)take(sizeof(unsigned short) * DD * DD);
    unsigned short* w2lo = (unsigned short*)take(sizeof(unsigned short) * DD * DD);
    unsigned short* wlhi = (unsigned short*)take(sizeof(unsigned short) * DD * DD);
    unsigned short* wllo = (unsigned short*)take(sizeof(unsigned short) * DD * DD);

    const int TB = 256;
    const int GB = (NN + 31) / 32;
    const int CW = (DD * DD + TB - 1) / TB;

    // Pre-split weights into transposed packed f16 hi/lo (one pass each)
    k_convw<<<CW, TB, 0, stream>>>(W1, w1hi, w1lo);
    k_convw<<<CW, TB, 0, stream>>>(W2, w2hi, w2lo);
    k_convw<<<CW, TB, 0, stream>>>(Wl, wlhi, wllo);

    // CSR by dst (shared by both layers), deterministic
    k_zero_i32<<<(NN + TB - 1) / TB, TB, 0, stream>>>(cnt, NN);
    k_hist<<<(EE + TB - 1) / TB, TB, 0, stream>>>(dstv, cnt);
    k_scan<<<1, 1024, 0, stream>>>(cnt, offs, cur);
    k_scatter<<<(EE + TB - 1) / TB, TB, 0, stream>>>(srcv, dstv, cur, csr);
    k_sort<<<(NN + TB - 1) / TB, TB, 0, stream>>>(offs, csr);

    // Layer 1: h = x@W1 ; attention ; aggregate+ELU -> gbuf
    k_gemm_bias<<<GB, TB, 0, stream>>>(x, w1hi, w1lo, nullptr, hbuf, NN);
    k_alproj<<<(NN * HH + TB - 1) / TB, TB, 0, stream>>>(hbuf, as1, ad1, als, ald);
    k_attn<<<(NN * HH + TB - 1) / TB, TB, 0, stream>>>(offs, csr, als, ald, wbuf, denom);
    k_agg<<<(NN * 32 + TB - 1) / TB, TB, 0, stream>>>(offs, csr, wbuf, denom, hbuf, gbuf);

    // Layer 2: h = g1@W2 ; attention ; aggregate+ELU -> gbuf
    k_gemm_bias<<<GB, TB, 0, stream>>>(gbuf, w2hi, w2lo, nullptr, hbuf, NN);
    k_alproj<<<(NN * HH + TB - 1) / TB, TB, 0, stream>>>(hbuf, as2, ad2, als, ald);
    k_attn<<<(NN * HH + TB - 1) / TB, TB, 0, stream>>>(offs, csr, als, ald, wbuf, denom);
    k_agg<<<(NN * 32 + TB - 1) / TB, TB, 0, stream>>>(offs, csr, wbuf, denom, hbuf, gbuf);

    // Final: out = g2@Wl + bl
    k_gemm_bias<<<GB, TB, 0, stream>>>(gbuf, wlhi, wllo, bl, out, NN);
}
